// SlotModel_3204045603607
// MI455X (gfx1250) — matrix operand order repaired and checked by reading, not compiled, not run on hardware
//
#include <hip/hip_runtime.h>
#include <hip/hip_bf16.h>
#include <math.h>

// CDNA5 / gfx1250, wave32. WMMA f16 FFN hot loop (~34 GFLOP), weights
// pre-swizzled once into a f16 fragment image and DMA'd to LDS per block via
// the Tensor Data Mover (tensor_load_to_lds + s_wait_tensorcnt).

typedef __attribute__((ext_vector_type(16))) _Float16     v16h;
typedef __attribute__((ext_vector_type(8)))  float        v8f;
typedef __attribute__((ext_vector_type(4)))  unsigned int v4u;
typedef __attribute__((ext_vector_type(8)))  int          v8i;
typedef __attribute__((ext_vector_type(4)))  int          v4i;

#define HD   64      // HIDDEN_DIM
#define FD   128     // HIDDEN_DIM*2
#define LSEQ 512
#define NB   2048    // batch
#define KTOP 6       // min(NUM_PAIRS+2, L-3)

// Packed weight image: W1 fragments (8192 halves) + W2 fragments (8192 halves)
// + 320 floats (b1[128], b2[64], gamma[64], beta[64]) = 34048 bytes.
#define IMG_HALVES (2 * 8192 + 640)
#define IMG_BYTES  (IMG_HALVES * 2)
#define IMG_QW     (IMG_BYTES / 8)       // 4256 8-byte elements
#define CHUNKS_PER_BLOCK 4

#if __has_builtin(__builtin_amdgcn_tensor_load_to_lds)
#define HAVE_TDM 1
#endif

// ---------------------------------------------------------------------------
// Kernel 0: pack W1/W2 into WMMA B-fragment layout (f16) + biases, once.
// B fragment (f16, 32xK chunk x 16 cols): lane L holds col n = L&15,
// halves e=0..7 hold K = kbase + (L>>4)*8 + e, e=8..15 hold K = kbase + 16 + (L>>4)*8 + (e-8).
// ---------------------------------------------------------------------------
__global__ __launch_bounds__(256)
void pack_weights_kernel(const float* __restrict__ W1, const float* __restrict__ b1,
                         const float* __restrict__ W2, const float* __restrict__ b2,
                         const float* __restrict__ gamma, const float* __restrict__ beta,
                         _Float16* __restrict__ img)
{
    const int tid    = threadIdx.x + blockIdx.x * blockDim.x;
    const int stride = blockDim.x * gridDim.x;
    for (int i = tid; i < 8192; i += stride) {          // W1: [kc:2][nt:8][lane:32][e:16]
        int e  = i & 15;
        int ln = (i >> 4) & 31;
        int nt = (i >> 9) & 7;
        int kc = i >> 12;
        int k  = kc * 32 + (ln >> 4) * 8 + (e & 7) + (e >> 3) * 16;
        int n  = nt * 16 + (ln & 15);
        img[i] = (_Float16)W1[k * FD + n];
    }
    for (int i = tid; i < 8192; i += stride) {          // W2: [kc:4][nt:4][lane:32][e:16]
        int e  = i & 15;
        int ln = (i >> 4) & 31;
        int nt = (i >> 9) & 3;
        int kc = i >> 11;
        int k  = kc * 32 + (ln >> 4) * 8 + (e & 7) + (e >> 3) * 16;
        int n  = nt * 16 + (ln & 15);
        img[8192 + i] = (_Float16)W2[k * HD + n];
    }
    float* fb = (float*)(img + 16384);
    for (int i = tid; i < FD; i += stride) fb[i] = b1[i];
    for (int i = tid; i < HD; i += stride) {
        fb[FD + i]          = b2[i];
        fb[FD + HD + i]     = gamma[i];
        fb[FD + 2 * HD + i] = beta[i];
    }
}

// ---------------------------------------------------------------------------
// Kernel 1: for every row r in [0, B*L): h = LN(e + FFN(e)), emit ||h||_2.
// 128 threads = 4 waves; each wave owns a 16-row M-tile; 4 x 64-row chunks
// per block. Weights arrive in LDS via one TDM DMA per block.
// ---------------------------------------------------------------------------
__global__ __launch_bounds__(128)
void ffn_ln_norms_kernel(const int* __restrict__ seq,
                         const float* __restrict__ embed,
                         const _Float16* __restrict__ img,
                         float* __restrict__ norms)
{
    __shared__ __align__(32) _Float16 sW[IMG_HALVES];   // 34048 B: W1|W2|biases
    __shared__ __align__(32) _Float16 sT[4][16 * FD];   // 16 KB per-wave scratch
    __shared__ int sVoc[4][16];

    const int tid   = threadIdx.x;
    const int wave  = tid >> 5;
    const int lane  = tid & 31;
    const int mrow  = lane & 15;   // row within 16-row tile
    const int khalf = lane >> 4;   // which K half this lane carries

    if (tid == 0) __builtin_prefetch(embed, 0, 3);

    // ---- DMA the packed weight image into LDS (wave 0 only) ---------------
    if (tid < 32) {
#if defined(HAVE_TDM)
        const unsigned int       lds_off = (unsigned int)(uintptr_t)(void*)&sW[0];
        const unsigned long long ga      = (unsigned long long)(uintptr_t)img;
        // D# group0: count=1 | lds_addr | global_addr[56:0] | type=2
        v4u g0 = { 1u,
                   lds_off,
                   (unsigned int)(ga & 0xffffffffu),
                   (unsigned int)((ga >> 32) & 0x01ffffffu) | 0x80000000u };
        // D# group1: data_size=3 (8B); 1-D tensor/tile of IMG_QW elements.
        v8i g1 = { (int)(3u << 16),                              // mask=0, data_size=8B
                   (int)(((unsigned)IMG_QW & 0xffffu) << 16),    // tensor_dim0 lo
                   (int)(1u << 16),                              // tensor_dim0 hi=0, tensor_dim1=1
                   (int)((unsigned)IMG_QW << 16),                // tensor_dim1 hi=0, tile_dim0
                   0,                                            // tile_dim1=0, tile_dim2=0
                   (int)IMG_QW,                                  // tensor_dim0_stride lo
                   (int)(((unsigned)IMG_QW & 0xffffu) << 16),    // stride0 hi=0, stride1 lo
                   0 };
        v4i gz = { 0, 0, 0, 0 };
#if defined(__clang_major__) && (__clang_major__ >= 23)
        v8i gz8 = { 0, 0, 0, 0, 0, 0, 0, 0 };
        __builtin_amdgcn_tensor_load_to_lds(g0, g1, gz, gz, gz8, 0);
#else
        __builtin_amdgcn_tensor_load_to_lds(g0, g1, gz, gz, 0);
#endif
        __builtin_amdgcn_s_wait_tensorcnt((short)0);
#else
        for (int i = lane; i < IMG_QW; i += 32)
            ((unsigned long long*)sW)[i] = ((const unsigned long long*)img)[i];
#endif
    }
    __syncthreads();

    const _Float16* sW1 = sW;               // 8192 halves
    const _Float16* sW2 = sW + 8192;        // 8192 halves
    const float*    fb  = (const float*)(sW + 16384);
    const float*    sB1 = fb;
    const float*    sB2 = fb + FD;
    const float*    sG  = fb + FD + HD;
    const float*    sBt = fb + FD + 2 * HD;

    for (int ch = 0; ch < CHUNKS_PER_BLOCK; ++ch) {
        const size_t row0 = ((size_t)blockIdx.x * CHUNKS_PER_BLOCK + ch) * 64
                            + (size_t)wave * 16;
        const size_t row  = row0 + mrow;
        const int    voc  = seq[row];
        if (khalf == 0) sVoc[wave][mrow] = voc;

        // ---- A-fragments of e (16x64 f16): ISA 16-bit A layout ------------
        // elements 0..7  -> K = kbase +      khalf*8 + e
        // elements 8..15 -> K = kbase + 16 + khalf*8 + e
        const float* erow = embed + (size_t)voc * HD;
        v16h a0, a1;
#pragma unroll
        for (int e = 0; e < 8; ++e) {
            a0[e]     = (_Float16)erow[ 0 + khalf * 8 + e];
            a0[e + 8] = (_Float16)erow[16 + khalf * 8 + e];
            a1[e]     = (_Float16)erow[32 + khalf * 8 + e];
            a1[e + 8] = (_Float16)erow[48 + khalf * 8 + e];
        }

        // ---- GEMM 1: t = relu(e @ W1 + b1), 16x128 per wave ---------------
#pragma unroll
        for (int nt = 0; nt < 8; ++nt) {
            v8f c = {};
            v16h bf0 = *(const v16h*)&sW1[((0 * 8 + nt) * 32 + lane) * 16];
            c = __builtin_amdgcn_wmma_f32_16x16x32_f16(false, a0, false, bf0, (short)0, c, false, false);
            v16h bf1 = *(const v16h*)&sW1[((1 * 8 + nt) * 32 + lane) * 16];
            c = __builtin_amdgcn_wmma_f32_16x16x32_f16(false, a1, false, bf1, (short)0, c, false, false);
            const int n    = nt * 16 + (lane & 15);
            const float bv = sB1[n];
#pragma unroll
            for (int r = 0; r < 8; ++r) {  // C tile: VGPR r -> row r + 8*khalf, col n
                float v = c[r] + bv;
                v = v > 0.f ? v : 0.f;
                sT[wave][(r + 8 * khalf) * FD + n] = (_Float16)v;
            }
        }
        __syncthreads();

        // ---- reload t as A-fragments (K = 128 -> 4 chunks) ----------------
        v16h ta[4];
#pragma unroll
        for (int kc = 0; kc < 4; ++kc) {
            const _Float16* tr = &sT[wave][mrow * FD + kc * 32 + khalf * 8];
#pragma unroll
            for (int e = 0; e < 8; ++e) { ta[kc][e] = tr[e]; ta[kc][e + 8] = tr[16 + e]; }
        }
        __syncthreads();

        // ---- GEMM 2: u = t @ W2 + b2, then x = e + u (16x64 per wave) -----
        v8f u[4];
#pragma unroll
        for (int nt = 0; nt < 4; ++nt) {
            v8f c = {};
#pragma unroll
            for (int kc = 0; kc < 4; ++kc) {
                v16h bf = *(const v16h*)&sW2[((kc * 4 + nt) * 32 + lane) * 16];
                c = __builtin_amdgcn_wmma_f32_16x16x32_f16(false, ta[kc], false, bf, (short)0, c, false, false);
            }
            u[nt] = c;
        }
        __syncthreads();

        float* xw = (float*)&sT[wave][0];   // reuse 4 KB scratch as f32 16x64
#pragma unroll
        for (int nt = 0; nt < 4; ++nt) {
            const int n    = nt * 16 + (lane & 15);
            const float bv = sB2[n];
#pragma unroll
            for (int r = 0; r < 8; ++r) {
                const int mr = r + 8 * khalf;
                const float ev = embed[(size_t)sVoc[wave][mr] * HD + n];
                xw[mr * HD + n] = u[nt][r] + bv + ev;
            }
        }
        __syncthreads();

        // ---- LayerNorm + L2 norm: 2 lanes per row -------------------------
        {
            const int pr = lane >> 1;           // row 0..15
            const int ph = lane & 1;            // half 0/1 (32 elems each)
            const float* xr = xw + pr * HD + ph * 32;
            float s = 0.f, s2 = 0.f;
#pragma unroll
            for (int j = 0; j < 32; ++j) { float v = xr[j]; s += v; s2 += v * v; }
            s  += __shfl_xor(s, 1, 32);
            s2 += __shfl_xor(s2, 1, 32);
            const float mu  = s * (1.f / 64.f);
            const float var = s2 * (1.f / 64.f) - mu * mu;
            const float rs  = rsqrtf(var + 1e-5f);
            float hs = 0.f;
#pragma unroll
            for (int j = 0; j < 32; ++j) {
                const int cidx = ph * 32 + j;
                const float h = (xr[j] - mu) * rs * sG[cidx] + sBt[cidx];
                hs += h * h;
            }
            hs += __shfl_xor(hs, 1, 32);
            if (ph == 0) norms[row0 + pr] = sqrtf(hs);
        }
        __syncthreads();   // before next chunk reuses sT / sVoc
    }
}

// ---------------------------------------------------------------------------
// Kernel 2: per batch -> top-6 over norms[0..508], recompute those h rows +
// the query row, attention over 6 slots, output projection. 64 threads/block.
// ---------------------------------------------------------------------------
__device__ __forceinline__ float blk_sum64(float v, float* red) {
    const int tid = threadIdx.x;
    red[tid] = v; __syncthreads();
#pragma unroll
    for (int off = 32; off > 0; off >>= 1) {
        if (tid < off) red[tid] += red[tid + off];
        __syncthreads();
    }
    float r = red[0];
    __syncthreads();
    return r;
}

__global__ __launch_bounds__(64)
void slot_head_kernel(const int* __restrict__ seq,
                      const float* __restrict__ embed,
                      const float* __restrict__ W1, const float* __restrict__ b1,
                      const float* __restrict__ W2, const float* __restrict__ b2,
                      const float* __restrict__ gamma, const float* __restrict__ beta,
                      const float* __restrict__ Wq, const float* __restrict__ bq,
                      const float* __restrict__ Wo, const float* __restrict__ bo,
                      const float* __restrict__ norms,
                      float* __restrict__ out)
{
    __shared__ float red[64];
    __shared__ int   redi[64];
    __shared__ int   topIdx[KTOP];
    __shared__ float slots[KTOP + 1][HD];    // 6 slots + query row h
    __shared__ float sE[HD], sTt[FD], sQ[HD], sCtx[HD], sScore[8];

    const int tid = threadIdx.x;
    const int b   = blockIdx.x;
    const float* nb = norms + (size_t)b * LSEQ;

    // ---- top-6 (descending, lowest index on ties) -------------------------
    for (int round = 0; round < KTOP; ++round) {
        float best = -1.f; int bi = -1;
        for (int l = tid; l < LSEQ - 3; l += 64) {
            bool skip = false;
            for (int r = 0; r < round; ++r) if (topIdx[r] == l) skip = true;
            const float v = nb[l];
            if (!skip && (v > best || (v == best && (unsigned)l < (unsigned)bi))) { best = v; bi = l; }
        }
        red[tid] = best; redi[tid] = bi; __syncthreads();
#pragma unroll
        for (int off = 32; off > 0; off >>= 1) {
            if (tid < off) {
                const float v2 = red[tid + off]; const int i2 = redi[tid + off];
                if (v2 > red[tid] || (v2 == red[tid] && (unsigned)i2 < (unsigned)redi[tid])) {
                    red[tid] = v2; redi[tid] = i2;
                }
            }
            __syncthreads();
        }
        if (tid == 0) topIdx[round] = redi[0];
        __syncthreads();
    }

    // ---- recompute h rows: 6 slots + query (l = L-1) ----------------------
    for (int s = 0; s <= KTOP; ++s) {
        const int l = (s < KTOP) ? topIdx[s] : (LSEQ - 1);
        const size_t grow = (size_t)b * LSEQ + (size_t)l;
        const int v = seq[grow];
        sE[tid] = embed[(size_t)v * HD + tid];
        __syncthreads();
        for (int j = tid; j < FD; j += 64) {
            float acc = b1[j];
            for (int k = 0; k < HD; ++k) acc += sE[k] * W1[k * FD + j];
            sTt[j] = fmaxf(acc, 0.f);
        }
        __syncthreads();
        float acc = b2[tid];
        for (int j = 0; j < FD; ++j) acc += sTt[j] * W2[j * HD + tid];
        const float x = sE[tid] + acc;
        __syncthreads();
        const float mu = blk_sum64(x, red) * (1.f / 64.f);
        const float d  = x - mu;
        const float var = blk_sum64(d * d, red) * (1.f / 64.f);
        const float rs  = rsqrtf(var + 1e-5f);
        slots[s][tid] = d * rs * gamma[tid] + beta[tid];
        __syncthreads();
    }

    // ---- attention head ---------------------------------------------------
    {
        float q = bq[tid];
        for (int k = 0; k < HD; ++k) q += slots[KTOP][k] * Wq[k * HD + tid];
        sQ[tid] = q;
    }
    __syncthreads();
    if (tid < KTOP) {
        float sc = 0.f;
        for (int k = 0; k < HD; ++k) sc += sQ[k] * slots[tid][k];
        sScore[tid] = sc * 0.125f;   // / sqrt(64)
    }
    __syncthreads();
    if (tid == 0) {
        float mx = sScore[0];
        for (int k = 1; k < KTOP; ++k) mx = fmaxf(mx, sScore[k]);
        float den = 0.f;
        for (int k = 0; k < KTOP; ++k) { const float e = expf(sScore[k] - mx); sScore[k] = e; den += e; }
        const float inv = 1.f / den;
        for (int k = 0; k < KTOP; ++k) sScore[k] *= inv;
    }
    __syncthreads();
    {
        float ctx = 0.f;
        for (int k = 0; k < KTOP; ++k) ctx += sScore[k] * slots[k][tid];
        sCtx[tid] = ctx;
    }
    __syncthreads();
    {
        float o = bo[tid];
        for (int k = 0; k < HD; ++k) o += sCtx[k] * Wo[k * HD + tid];
        out[(size_t)b * HD + tid] = o;
    }
}

// ---------------------------------------------------------------------------
extern "C" void kernel_launch(void* const* d_in, const int* in_sizes, int n_in,
                              void* d_out, int out_size, void* d_ws, size_t ws_size,
                              hipStream_t stream)
{
    const int*   seq   = (const int*)  d_in[0];
    const float* embed = (const float*)d_in[1];
    const float* W1    = (const float*)d_in[2];
    const float* b1    = (const float*)d_in[3];
    const float* W2    = (const float*)d_in[4];
    const float* b2    = (const float*)d_in[5];
    const float* gamma = (const float*)d_in[6];
    const float* beta  = (const float*)d_in[7];
    const float* Wq    = (const float*)d_in[8];
    const float* bq    = (const float*)d_in[9];
    const float* Wo    = (const float*)d_in[10];
    const float* bo    = (const float*)d_in[11];

    _Float16* img   = (_Float16*)d_ws;                      // 34048 B packed weights
    float*    norms = (float*)((char*)d_ws + 65536);        // B * 512 floats = 4 MB
    float*    outp  = (float*)d_out;                        // B * 64 f32

    pack_weights_kernel<<<32, 256, 0, stream>>>(W1, b1, W2, b2, gamma, beta, img);

    const int totalRows = NB * LSEQ;                        // 1,048,576
    ffn_ln_norms_kernel<<<totalRows / (64 * CHUNKS_PER_BLOCK), 128, 0, stream>>>(
        seq, embed, img, norms);

    slot_head_kernel<<<NB, 64, 0, stream>>>(
        seq, embed, W1, b1, W2, b2, gamma, beta, Wq, bq, Wo, bo, norms, outp);
}
